// Cleaner_37254546325588
// MI455X (gfx1250) — matrix operand order repaired and checked by reading, not compiled, not run on hardware
//
#include <hip/hip_runtime.h>
#include <hip/hip_bf16.h>

typedef __attribute__((ext_vector_type(16))) _Float16 v16h;
typedef __attribute__((ext_vector_type(8)))  _Float16 v8h;
typedef __attribute__((ext_vector_type(8)))  float    v8f;

union AFrag { v16h v; v8h h[2]; };

struct Segs {
  const _Float16* ptr[6];
  int width[6];   // in halfs; always a multiple of 32
  int kind[6];    // 0 = gather by row[], 1 = gather by col[], 2 = per-edge
  int nseg;
  int fin;        // total input width (multiple of 32)
};

#define BM 32   // edges per workgroup

// ---------------- small utility kernels ----------------

__global__ __launch_bounds__(256) void gnn_to_half(const float* __restrict__ x,
                                                   _Float16* __restrict__ y, int n) {
  int i = blockIdx.x * 256 + threadIdx.x;
  if (i < n) y[i] = (_Float16)x[i];
}

// Wt[n][k] = (f16) W[k][n]   (W is [fin][nout] row-major)
__global__ __launch_bounds__(256) void gnn_wt_convert(const float* __restrict__ W,
                                                      _Float16* __restrict__ Wt,
                                                      int fin, int nout) {
  int i = blockIdx.x * 256 + threadIdx.x;
  int total = fin * nout;
  if (i < total) {
    int n = i / fin;
    int k = i - n * fin;
    Wt[i] = (_Float16)W[(size_t)k * nout + n];
  }
}

__global__ __launch_bounds__(256) void gnn_fill_f32(float* __restrict__ p, float v, int n) {
  int i = blockIdx.x * 256 + threadIdx.x;
  if (i < n) p[i] = v;
}

__global__ __launch_bounds__(256) void gnn_degree(const int* __restrict__ row,
                                                  float* __restrict__ deg, int E) {
  int e = blockIdx.x * 256 + threadIdx.x;
  if (e < E)
    __hip_atomic_fetch_add(&deg[row[e]], 1.0f, __ATOMIC_RELAXED, __HIP_MEMORY_SCOPE_AGENT);
}

// x = relu(xacc / max(deg,1)), to f16 buffer or f32 (final layer)
__global__ __launch_bounds__(256) void gnn_finalize(const float* __restrict__ xacc,
                                                    const float* __restrict__ deg,
                                                    _Float16* __restrict__ xh,
                                                    float* __restrict__ xf, int N) {
  int i = blockIdx.x * 256 + threadIdx.x;
  if (i < N * 128) {
    float d = deg[i >> 7];
    d = d > 1.f ? d : 1.f;
    float v = xacc[i] / d;
    v = v > 0.f ? v : 0.f;
    if (xh) xh[i] = (_Float16)v;
    else    xf[i] = v;
  }
}

// out[e] = ep[e,:] . wl + bl
__global__ __launch_bounds__(256) void gnn_final_lin(const float* __restrict__ ep,
                                                     const float* __restrict__ wl,
                                                     const float* __restrict__ bl,
                                                     float* __restrict__ out, int E) {
  int e = blockIdx.x * 256 + threadIdx.x;
  if (e < E) {
    const float4* p = (const float4*)(ep + (size_t)e * 128);
    const float4* w = (const float4*)wl;
    float s = bl[0];
#pragma unroll
    for (int i = 0; i < 32; ++i) {
      float4 a = p[i];
      float4 b = w[i];
      s += a.x * b.x + a.y * b.y + a.z * b.z + a.w * b.w;
    }
    out[e] = s;
  }
}

// ---------------- fused per-edge 2-layer MLP (WMMA) ----------------
// 32 edges x 128 outputs per workgroup; 8 waves, each wave owns one 16-edge
// m-tile and two 16-col n-tiles. Stage1: h1 = relu(A@W1+b1) staged through LDS
// as f16; Stage2: h = h1@W2+b2. relu(h) -> e_out, pre-relu h atomically
// scattered into xacc[row[e]].
__global__ __launch_bounds__(256) void gnn_edge_mlp(
    Segs segs, const int* __restrict__ row, const int* __restrict__ col,
    const _Float16* __restrict__ W1t, const float* __restrict__ b1,
    const _Float16* __restrict__ W2t, const float* __restrict__ b2,
    _Float16* __restrict__ eout_h, float* __restrict__ eout_f,
    float* __restrict__ xacc) {
  __shared__ _Float16 Af[BM * 768];   // gathered input rows (max fin = 768)
  __shared__ _Float16 T[BM * 128];    // stage-1 activations
  __shared__ int sRow[BM], sCol[BM];

  const int tid = threadIdx.x;
  const int e0 = blockIdx.x * BM;
  if (tid < BM) {
    sRow[tid] = row[e0 + tid];
    sCol[tid] = col[e0 + tid];
  }
  __syncthreads();

  // ---- stage gathered A rows into LDS, 16B chunks ----
  const int fin = segs.fin;
  const int cpr = fin >> 3;            // 8-half chunks per row
  const int total = BM * cpr;
  for (int idx = tid; idx < total; idx += 256) {
    int m = idx / cpr;
    int k = (idx - m * cpr) << 3;      // half offset within concat row
    int s = 0, kk = k;
    while (kk >= segs.width[s]) { kk -= segs.width[s]; ++s; }
    int kd = segs.kind[s];
    int r;
    if (kd == 0)      r = sRow[m];
    else if (kd == 1) r = sCol[m];
    else              r = e0 + m;
    const v8h* src = (const v8h*)(segs.ptr[s] + (size_t)r * segs.width[s] + kk);
    *(v8h*)&Af[m * fin + k] = *src;
  }
  __syncthreads();

  const int lane = tid & 31;
  const int wave = tid >> 5;
  const int nl = lane & 15;
  const int hi = lane >> 4;            // 0: K low half, 1: K high half
  const int mt = wave & 1;             // m-tile (0/1)
  const int n0 = ((wave >> 1) * 2) * 16 + nl;  // first n-tile column
  const int n1 = n0 + 16;                      // second n-tile column
  const int am = mt * 16 + nl;         // A-matrix row for this lane

  // ---- stage 1: [32 x fin] @ [fin x 128] ----
  v8f acc0 = {}; v8f acc1 = {};
  for (int k0 = 0; k0 < fin; k0 += 32) {
    AFrag a;
    a.h[0] = *(const v8h*)&Af[am * fin + k0 + hi * 16];
    a.h[1] = *(const v8h*)&Af[am * fin + k0 + hi * 16 + 8];
    v16h bf0 = *(const v16h*)(W1t + (size_t)n0 * fin + k0 + hi * 16);
    v16h bf1 = *(const v16h*)(W1t + (size_t)n1 * fin + k0 + hi * 16);
    acc0 = __builtin_amdgcn_wmma_f32_16x16x32_f16(false, a.v, false, bf0, (short)0, acc0, false, false);
    acc1 = __builtin_amdgcn_wmma_f32_16x16x32_f16(false, a.v, false, bf1, (short)0, acc1, false, false);
  }

  // bias + relu + f16, store to LDS in row-major [m][n]
  {
    float bb0 = b1[n0], bb1 = b1[n1];
#pragma unroll
    for (int r = 0; r < 8; ++r) {
      int mr = mt * 16 + r + hi * 8;   // C/D layout: m = r + 8*hi
      float v0 = acc0[r] + bb0; v0 = v0 > 0.f ? v0 : 0.f;
      float v1 = acc1[r] + bb1; v1 = v1 > 0.f ? v1 : 0.f;
      T[mr * 128 + n0] = (_Float16)v0;
      T[mr * 128 + n1] = (_Float16)v1;
    }
  }
  __syncthreads();

  // ---- stage 2: [32 x 128] @ [128 x 128] ----
  v8f h0 = {}; v8f h1 = {};
  for (int k0 = 0; k0 < 128; k0 += 32) {
    AFrag a;
    a.h[0] = *(const v8h*)&T[am * 128 + k0 + hi * 16];
    a.h[1] = *(const v8h*)&T[am * 128 + k0 + hi * 16 + 8];
    v16h bf0 = *(const v16h*)(W2t + (size_t)n0 * 128 + k0 + hi * 16);
    v16h bf1 = *(const v16h*)(W2t + (size_t)n1 * 128 + k0 + hi * 16);
    h0 = __builtin_amdgcn_wmma_f32_16x16x32_f16(false, a.v, false, bf0, (short)0, h0, false, false);
    h1 = __builtin_amdgcn_wmma_f32_16x16x32_f16(false, a.v, false, bf1, (short)0, h1, false, false);
  }

  // ---- epilogue: e_out = relu(h); scatter pre-relu h to xacc[row] ----
  float bb0 = b2[n0], bb1 = b2[n1];
#pragma unroll
  for (int r = 0; r < 8; ++r) {
    int mr = mt * 16 + r + hi * 8;
    int e = e0 + mr;
    int nr = sRow[mr];
    float v0 = h0[r] + bb0;
    float v1 = h1[r] + bb1;
    __hip_atomic_fetch_add(&xacc[(size_t)nr * 128 + n0], v0, __ATOMIC_RELAXED, __HIP_MEMORY_SCOPE_AGENT);
    __hip_atomic_fetch_add(&xacc[(size_t)nr * 128 + n1], v1, __ATOMIC_RELAXED, __HIP_MEMORY_SCOPE_AGENT);
    float r0 = v0 > 0.f ? v0 : 0.f;
    float r1 = v1 > 0.f ? v1 : 0.f;
    if (eout_h) {
      eout_h[(size_t)e * 128 + n0] = (_Float16)r0;
      eout_h[(size_t)e * 128 + n1] = (_Float16)r1;
    } else {
      eout_f[(size_t)e * 128 + n0] = r0;
      eout_f[(size_t)e * 128 + n1] = r1;
    }
  }
}

// ---------------- host orchestration ----------------

extern "C" void kernel_launch(void* const* d_in, const int* in_sizes, int n_in,
                              void* d_out, int out_size, void* d_ws, size_t ws_size,
                              hipStream_t stream) {
  (void)in_sizes; (void)n_in; (void)out_size; (void)ws_size;
  const int N = 30000, E = 480000;

  const float* node_feat = (const float*)d_in[0];
  const float* edge_feat = (const float*)d_in[1];
  const int*   eidx      = (const int*)d_in[2];
  const int* row = eidx;
  const int* col = eidx + E;

  const float *W1[6], *B1[6], *W2[6], *B2[6];
  for (int l = 0; l < 6; ++l) {
    W1[l] = (const float*)d_in[3 + 4 * l];
    B1[l] = (const float*)d_in[4 + 4 * l];
    W2[l] = (const float*)d_in[5 + 4 * l];
    B2[l] = (const float*)d_in[6 + 4 * l];
  }
  const float* Wlin = (const float*)d_in[27];
  const float* Blin = (const float*)d_in[28];
  const int fin1[6] = {160, 384, 768, 768, 768, 768};

  // workspace carve-out
  char* ws = (char*)d_ws;
  size_t off = 0;
  auto walloc = [&](size_t bytes) -> void* {
    void* p = ws + off;
    off += (bytes + 255) & ~(size_t)255;
    return p;
  };
  _Float16* nf16 = (_Float16*)walloc((size_t)N * 64 * 2);
  _Float16* ef16 = (_Float16*)walloc((size_t)E * 32 * 2);
  _Float16* xb[3];
  for (int i = 0; i < 3; ++i) xb[i] = (_Float16*)walloc((size_t)N * 128 * 2);
  _Float16* ebuf[3];
  for (int i = 0; i < 3; ++i) ebuf[i] = (_Float16*)walloc((size_t)E * 128 * 2);
  float* xacc = (float*)walloc((size_t)N * 128 * 4);
  float* deg  = (float*)walloc((size_t)N * 4);
  _Float16 *W1t[6], *W2t[6];
  for (int l = 0; l < 6; ++l) {
    W1t[l] = (_Float16*)walloc((size_t)128 * fin1[l] * 2);
    W2t[l] = (_Float16*)walloc((size_t)128 * 128 * 2);
  }

  // precompute: f16 features, transposed f16 weights, degrees
  gnn_to_half<<<(N * 64 + 255) / 256, 256, 0, stream>>>(node_feat, nf16, N * 64);
  gnn_to_half<<<(E * 32 + 255) / 256, 256, 0, stream>>>(edge_feat, ef16, E * 32);
  for (int l = 0; l < 6; ++l) {
    int t1 = fin1[l] * 128;
    gnn_wt_convert<<<(t1 + 255) / 256, 256, 0, stream>>>(W1[l], W1t[l], fin1[l], 128);
    gnn_wt_convert<<<(128 * 128 + 255) / 256, 256, 0, stream>>>(W2[l], W2t[l], 128, 128);
  }
  gnn_fill_f32<<<(N + 255) / 256, 256, 0, stream>>>(deg, 0.f, N);
  gnn_degree<<<(E + 255) / 256, 256, 0, stream>>>(row, deg, E);

  // d_out tuple layout: out [E], xp [N*128], ep [E*128]
  float* out_lin = (float*)d_out;
  float* out_xp  = out_lin + E;
  float* out_ep  = out_xp + (size_t)N * 128;

  auto run_conv = [&](const Segs& sg, int l, _Float16* eh, float* ef,
                      _Float16* xh, float* xf) {
    gnn_fill_f32<<<(N * 128 + 255) / 256, 256, 0, stream>>>(xacc, 0.f, N * 128);
    gnn_edge_mlp<<<E / BM, 256, 0, stream>>>(sg, row, col, W1t[l], B1[l],
                                             W2t[l], B2[l], eh, ef, xacc);
    gnn_finalize<<<(N * 128 + 255) / 256, 256, 0, stream>>>(xacc, deg, xh, xf, N);
  };

  Segs sg{};
  // conv1: [x0[row] | x0[col] | e0], fin = 160
  sg = Segs{};
  sg.ptr[0] = nf16; sg.width[0] = 64; sg.kind[0] = 0;
  sg.ptr[1] = nf16; sg.width[1] = 64; sg.kind[1] = 1;
  sg.ptr[2] = ef16; sg.width[2] = 32; sg.kind[2] = 2;
  sg.nseg = 3; sg.fin = 160;
  run_conv(sg, 0, ebuf[0], nullptr, xb[0], nullptr);

  // conv2: [x1[row] | x1[col] | e1], fin = 384
  sg = Segs{};
  sg.ptr[0] = xb[0];   sg.width[0] = 128; sg.kind[0] = 0;
  sg.ptr[1] = xb[0];   sg.width[1] = 128; sg.kind[1] = 1;
  sg.ptr[2] = ebuf[0]; sg.width[2] = 128; sg.kind[2] = 2;
  sg.nseg = 3; sg.fin = 384;
  run_conv(sg, 1, ebuf[1], nullptr, xb[1], nullptr);

  // conv3..6: [xp[row]|xpp[row] | xp[col]|xpp[col] | ep|epp], fin = 768
  int xpI = 1, xppI = 0, epI = 1, eppI = 0, outI = 2;
  for (int l = 2; l < 6; ++l) {
    sg = Segs{};
    sg.ptr[0] = xb[xpI];    sg.width[0] = 128; sg.kind[0] = 0;
    sg.ptr[1] = xb[xppI];   sg.width[1] = 128; sg.kind[1] = 0;
    sg.ptr[2] = xb[xpI];    sg.width[2] = 128; sg.kind[2] = 1;
    sg.ptr[3] = xb[xppI];   sg.width[3] = 128; sg.kind[3] = 1;
    sg.ptr[4] = ebuf[epI];  sg.width[4] = 128; sg.kind[4] = 2;
    sg.ptr[5] = ebuf[eppI]; sg.width[5] = 128; sg.kind[5] = 2;
    sg.nseg = 6; sg.fin = 768;
    if (l < 5) {
      run_conv(sg, l, ebuf[outI], nullptr, xb[outI], nullptr);
      xppI = xpI; xpI = outI;
      eppI = epI; epI = outI;
      outI = (outI + 1) % 3;
    } else {
      // conv6: write xp / ep directly as fp32 into d_out tuple regions
      run_conv(sg, l, nullptr, out_ep, nullptr, out_xp);
    }
  }

  gnn_final_lin<<<(E + 255) / 256, 256, 0, stream>>>(out_ep, Wlin, Blin, out_lin, E);
}
